// ImageCaptioningTransformer_90649579750161
// MI455X (gfx1250) — matrix operand order repaired and checked by reading, not compiled, not run on hardware
//
#include <hip/hip_runtime.h>
#include <hip/hip_bf16.h>
#include <cstdint>
#include <cstddef>

#define B_  4
#define S_  512
#define H_  768
#define NH_ 8
#define HD_ 96
#define L_  6
#define V_  50257
#define T_  50
#define FF_ 3072
#define BS_ (B_ * S_)   // 2048 token rows
#define BT_ (B_ * T_)   // 200 image rows

// LDS tile strides (halves). 40 = 32 + 8 pad: keeps 16-B alignment for b128
// access while spreading bank mapping (stride 80 B -> 20 banks, gcd(20,64)=4).
#define ASTR_ 40
#define BSTR_ 40

// ---------------------------------------------------------------------------
// CDNA5 WMMA fragment types (wave32): 16x16x32 bf16, fp32 accumulate
// ---------------------------------------------------------------------------
typedef __attribute__((ext_vector_type(16))) __bf16 v16bf;
typedef __attribute__((ext_vector_type(8)))  float  v8f;

union U16x8 { uint32_t u[8]; uint4 q[2]; v16bf v; };

__device__ __forceinline__ uint16_t f2bf(float f) {
  uint32_t u = __float_as_uint(f);
  u += 0x7FFFu + ((u >> 16) & 1u);   // round-to-nearest-even
  return (uint16_t)(u >> 16);
}
__device__ __forceinline__ float bf2f(uint16_t h) {
  return __uint_as_float(((uint32_t)h) << 16);
}
__device__ __forceinline__ float bfLo(uint32_t w) { return __uint_as_float(w << 16); }
__device__ __forceinline__ float bfHi(uint32_t w) { return __uint_as_float(w & 0xffff0000u); }

// ---------------------------------------------------------------------------
// Elementwise helpers
// ---------------------------------------------------------------------------
__global__ void cvt_bf16_kernel(const float* __restrict__ src,
                                uint16_t* __restrict__ dst,
                                long long n, float scale) {
  long long i = (long long)blockIdx.x * blockDim.x + threadIdx.x;
  if (i < n) dst[i] = f2bf(src[i] * scale);
}

__global__ void scale_f32_kernel(const float* __restrict__ src,
                                 float* __restrict__ dst,
                                 long long n, float s) {
  long long i = (long long)blockIdx.x * blockDim.x + threadIdx.x;
  if (i < n) dst[i] = src[i] * s;
}

__global__ void embed_kernel(const int* __restrict__ tok,
                             const float* __restrict__ tokEmb,
                             const float* __restrict__ posEmb,
                             float* __restrict__ x) {
  int i  = blockIdx.x * 256 + threadIdx.x;     // n = BS_*H_ (multiple of 256)
  int rs = i / H_;
  int c  = i - rs * H_;
  int s  = rs % S_;
  x[i] = tokEmb[(size_t)tok[rs] * H_ + c] + posEmb[(size_t)s * H_ + c];
}

// ---------------------------------------------------------------------------
// LayerNorm: one wave per row (768 cols = 24 per lane), bf16 output
// ---------------------------------------------------------------------------
__global__ void __launch_bounds__(128) ln_kernel(
    const float* __restrict__ x, const float* __restrict__ g,
    const float* __restrict__ bta, uint16_t* __restrict__ y, int rows) {
  const int lane = threadIdx.x & 31;
  const int wave = threadIdx.x >> 5;
  const int row  = blockIdx.x * 4 + wave;
  if (row >= rows) return;
  const float* xr = x + (size_t)row * H_;
  float v[H_ / 32];
  float s = 0.f;
#pragma unroll
  for (int i = 0; i < H_ / 32; ++i) { v[i] = xr[lane + 32 * i]; s += v[i]; }
#pragma unroll
  for (int m = 16; m >= 1; m >>= 1) s += __shfl_xor(s, m, 32);
  float mean = s * (1.0f / H_);
  float var = 0.f;
#pragma unroll
  for (int i = 0; i < H_ / 32; ++i) { float d = v[i] - mean; var += d * d; }
#pragma unroll
  for (int m = 16; m >= 1; m >>= 1) var += __shfl_xor(var, m, 32);
  float inv = rsqrtf(var * (1.0f / H_) + 1e-5f);
  uint16_t* yr = y + (size_t)row * H_;
#pragma unroll
  for (int i = 0; i < H_ / 32; ++i) {
    int c = lane + 32 * i;
    yr[c] = f2bf((v[i] - mean) * inv * g[c] + bta[c]);
  }
}

// ---------------------------------------------------------------------------
// Tiled WMMA GEMM: C[M,N] = act(A[M,K] @ B[K,N] + bias) (+ residual)
//   A, B bf16 row-major globally; fp32 accumulate, v_wmma_f32_16x16x32_bf16.
//   Block = 256 threads = 8 waves; workgroup tile 128x128; each wave 32x64
//   (2x4 wmma tiles). K staged in LDS in chunks of 32.
//   LDS A tile: row-major [128][ASTR_]   -> A fragment = 2x ds_load_b128.
//   LDS B tile: col-major pair-packed [128 cols][BSTR_ halves of K]
//               -> B fragment = 2x ds_load_b128, no VALU repacking.
// ---------------------------------------------------------------------------
__global__ void __launch_bounds__(256) gemm_bf16_wmma(
    const uint16_t* __restrict__ A, const uint16_t* __restrict__ Bm,
    const float* __restrict__ bias, const float* __restrict__ resid,
    float* __restrict__ outF, uint16_t* __restrict__ outB,
    int M, int N, int K, int act) {
  __shared__ uint16_t lA[128 * ASTR_];
  __shared__ uint16_t lB[128 * BSTR_];

  const int tid   = threadIdx.x;
  const int lane  = tid & 31;
  const int wave  = tid >> 5;
  const int waveM = wave & 3;     // 4 wave-rows of 32
  const int waveN = wave >> 2;    // 2 wave-cols of 64
  const int m0 = blockIdx.y * 128;
  const int n0 = blockIdx.x * 128;
  const int half = lane >> 4;     // 0: lanes 0-15, 1: lanes 16-31
  const int sub  = lane & 15;
  const bool nFast = (N & 7) == 0;

  v8f acc[2][4] = {};

  for (int kt = 0; kt < K; kt += 32) {
    // ---- A tile 128x32 bf16: 512 16-byte chunks, 2 per thread -------------
#pragma unroll
    for (int i = 0; i < 2; ++i) {
      int c = tid + i * 256;
      int row = c >> 2, seg = c & 3;
      uint4 val = make_uint4(0u, 0u, 0u, 0u);
      if (m0 + row < M)
        val = *reinterpret_cast<const uint4*>(A + (size_t)(m0 + row) * K + kt + seg * 8);
      *reinterpret_cast<uint4*>(&lA[row * ASTR_ + seg * 8]) = val;
    }
    // ---- B tile 32x128 bf16, stored transposed into lB[col][k] ------------
    if (nFast) {
      // N % 8 == 0: vector global loads (2x b128 per thread)
#pragma unroll
      for (int i = 0; i < 2; ++i) {
        int c2 = tid + i * 256;           // 512 chunks of 8 halves
        int r = c2 >> 4, cs = (c2 & 15) * 8;
        uint4 g = make_uint4(0u, 0u, 0u, 0u);
        if (n0 + cs < N)
          g = *reinterpret_cast<const uint4*>(Bm + (size_t)(kt + r) * N + n0 + cs);
        uint32_t ws[4] = {g.x, g.y, g.z, g.w};
#pragma unroll
        for (int j = 0; j < 4; ++j) {
          lB[(cs + 2 * j)     * BSTR_ + r] = (uint16_t)(ws[j] & 0xffffu);
          lB[(cs + 2 * j + 1) * BSTR_ + r] = (uint16_t)(ws[j] >> 16);
        }
      }
    } else {
      // odd N (vocab head): guarded scalar loads
#pragma unroll
      for (int i = 0; i < 16; ++i) {
        int hidx = i * 256 + tid;
        int r = hidx >> 7, c = hidx & 127;
        uint16_t val = 0;
        if (n0 + c < N) val = Bm[(size_t)(kt + r) * N + n0 + c];
        lB[c * BSTR_ + r] = val;
      }
    }
    __syncthreads();

    // ---- A fragments: lane = row sub, halves [half*8, +8) and [16+half*8, +8)
    U16x8 af[2];
#pragma unroll
    for (int mi = 0; mi < 2; ++mi) {
      const uint16_t* ar = &lA[(waveM * 32 + mi * 16 + sub) * ASTR_ + half * 8];
      af[mi].q[0] = *reinterpret_cast<const uint4*>(ar);
      af[mi].q[1] = *reinterpret_cast<const uint4*>(ar + 16);
    }
    // ---- B fragments: lane = col sub, halves [half*8, +8) and [16+half*8, +8) of that column
    U16x8 bfm[4];
#pragma unroll
    for (int ni = 0; ni < 4; ++ni) {
      const uint16_t* br = &lB[(waveN * 64 + ni * 16 + sub) * BSTR_ + half * 8];
      bfm[ni].q[0] = *reinterpret_cast<const uint4*>(br);
      bfm[ni].q[1] = *reinterpret_cast<const uint4*>(br + 16);
    }
#pragma unroll
    for (int mi = 0; mi < 2; ++mi)
#pragma unroll
      for (int ni = 0; ni < 4; ++ni)
        acc[mi][ni] = __builtin_amdgcn_wmma_f32_16x16x32_bf16(
            false, af[mi].v, false, bfm[ni].v, (short)0, acc[mi][ni], false, false);

    __syncthreads();
  }

  // Epilogue. C/D layout: VGPR r -> row = half*8 + r, col = sub.
#pragma unroll
  for (int mi = 0; mi < 2; ++mi)
    for (int ni = 0; ni < 4; ++ni)
#pragma unroll
      for (int r = 0; r < 8; ++r) {
        int row = m0 + waveM * 32 + mi * 16 + half * 8 + r;
        int col = n0 + waveN * 64 + ni * 16 + sub;
        if (row < M && col < N) {
          float v = acc[mi][ni][r];
          if (bias) v += bias[col];
          if (act)  v = 0.5f * v * (1.0f + erff(v * 0.70710678118654752f)); // exact GELU
          size_t idx = (size_t)row * N + col;
          if (resid) v += resid[idx];
          if (outF) outF[idx] = v;
          if (outB) outB[idx] = f2bf(v);
        }
      }
}

// ---------------------------------------------------------------------------
// Attention: one wave per query row, 8 rows per block. Scores + softmax in
// LDS (fp32), bf16 I/O. Shared by self (Skv=512) and cross (Skv=50).
// K rows read as b128 (12 per key), V as dwords (2 dims per load).
// ---------------------------------------------------------------------------
__global__ void __launch_bounds__(256) attn_kernel(
    const uint16_t* __restrict__ Q, int qStride,
    const uint16_t* __restrict__ KV, int kvStride, int kBase, int vBase,
    uint16_t* __restrict__ O, int Skv, float scale) {
  __shared__ float sQ[8][HD_];
  __shared__ float sP[8][S_];
  const int lane = threadIdx.x & 31;
  const int wave = threadIdx.x >> 5;
  const int bh = blockIdx.x;
  const int b = bh / NH_, h = bh % NH_;
  const int qrow = blockIdx.y * 8 + wave;
  const size_t qIdx = (size_t)(b * S_ + qrow) * qStride + h * HD_;

  for (int d = lane; d < HD_; d += 32) sQ[wave][d] = bf2f(Q[qIdx + d]);
  __syncthreads();

  // scores: each lane owns keys lane, lane+32, ...
  float mx = -3.0e38f;
  for (int kk = lane; kk < Skv; kk += 32) {
    const uint4* kr4 = reinterpret_cast<const uint4*>(
        KV + (size_t)(b * Skv + kk) * kvStride + kBase + h * HD_);
    float s = 0.f;
#pragma unroll
    for (int i = 0; i < HD_ / 8; ++i) {
      uint4 wv = kr4[i];
      const float* qp = &sQ[wave][i * 8];
      uint32_t ws[4] = {wv.x, wv.y, wv.z, wv.w};
#pragma unroll
      for (int j = 0; j < 4; ++j) {
        s += qp[2 * j]     * bfLo(ws[j]);
        s += qp[2 * j + 1] * bfHi(ws[j]);
      }
    }
    s *= scale;
    sP[wave][kk] = s;
    mx = fmaxf(mx, s);
  }
#pragma unroll
  for (int m = 16; m >= 1; m >>= 1) mx = fmaxf(mx, __shfl_xor(mx, m, 32));

  float sum = 0.f;
  for (int kk = lane; kk < Skv; kk += 32) {
    float e = __expf(sP[wave][kk] - mx);
    sP[wave][kk] = e;
    sum += e;
  }
#pragma unroll
  for (int m = 16; m >= 1; m >>= 1) sum += __shfl_xor(sum, m, 32);
  float inv = 1.0f / sum;
  __syncthreads();

  // output: lane owns dword w -> head dims 2w, 2w+1
  const size_t oBase = (size_t)(b * S_ + qrow) * H_ + h * HD_;
  for (int w = lane; w < HD_ / 2; w += 32) {
    float a0 = 0.f, a1 = 0.f;
    for (int kk = 0; kk < Skv; ++kk) {
      const uint32_t* vr = reinterpret_cast<const uint32_t*>(
          KV + (size_t)(b * Skv + kk) * kvStride + vBase + h * HD_);
      uint32_t vv = vr[w];
      float p = sP[wave][kk];
      a0 += p * bfLo(vv);
      a1 += p * bfHi(vv);
    }
    uint32_t packed = (uint32_t)f2bf(a0 * inv) | ((uint32_t)f2bf(a1 * inv) << 16);
    *reinterpret_cast<uint32_t*>(O + oBase + 2 * w) = packed;
  }
}

// ---------------------------------------------------------------------------
// Host orchestration
// ---------------------------------------------------------------------------
extern "C" void kernel_launch(void* const* d_in, const int* in_sizes, int n_in,
                              void* d_out, int out_size, void* d_ws, size_t ws_size,
                              hipStream_t stream) {
  (void)in_sizes; (void)n_in; (void)out_size; (void)ws_size;
  const int*   tok    = (const int*)  d_in[0];
  const float* img    = (const float*)d_in[1];
  const float* tokEmb = (const float*)d_in[2];
  const float* posEmb = (const float*)d_in[3];
  const float* sqkv_w = (const float*)d_in[4];
  const float* sqkv_b = (const float*)d_in[5];
  const float* so_w   = (const float*)d_in[6];
  const float* so_b   = (const float*)d_in[7];
  const float* cq_w   = (const float*)d_in[8];
  const float* cq_b   = (const float*)d_in[9];
  const float* ckv_w  = (const float*)d_in[10];
  const float* ckv_b  = (const float*)d_in[11];
  const float* co_w   = (const float*)d_in[12];
  const float* co_b   = (const float*)d_in[13];
  const float* ff1_w  = (const float*)d_in[14];
  const float* ff1_b  = (const float*)d_in[15];
  const float* ff2_w  = (const float*)d_in[16];
  const float* ff2_b  = (const float*)d_in[17];
  const float* ln1_g  = (const float*)d_in[18];
  const float* ln1_b  = (const float*)d_in[19];
  const float* ln2_g  = (const float*)d_in[20];
  const float* ln2_b  = (const float*)d_in[21];
  const float* ln3_g  = (const float*)d_in[22];
  const float* ln3_b  = (const float*)d_in[23];
  const float* out_w  = (const float*)d_in[24];
  const float* out_b  = (const float*)d_in[25];

  uint8_t* wsBase = (uint8_t*)d_ws;
  size_t off = 0;
  auto alloc  = [&](size_t bytes) -> void* {
    off = (off + 255) & ~(size_t)255;
    void* p = wsBase + off;
    off += bytes;
    return p;
  };
  auto allocH = [&](size_t elems) -> uint16_t* { return (uint16_t*)alloc(elems * 2); };
  auto allocF = [&](size_t elems) -> float*    { return (float*)   alloc(elems * 4); };

  // bf16 weight copies (~190 MB) + activations (~50 MB)
  uint16_t* wQKV = allocH((size_t)L_ * H_ * 3 * H_);
  uint16_t* wSO  = allocH((size_t)L_ * H_ * H_);
  uint16_t* wCQ  = allocH((size_t)L_ * H_ * H_);
  uint16_t* wCKV = allocH((size_t)L_ * H_ * 2 * H_);
  uint16_t* wCO  = allocH((size_t)L_ * H_ * H_);
  uint16_t* wF1  = allocH((size_t)L_ * H_ * FF_);
  uint16_t* wF2  = allocH((size_t)L_ * FF_ * H_);
  uint16_t* wOUT = allocH((size_t)H_ * V_);
  uint16_t* imgB = allocH((size_t)BT_ * H_);
  float*    xF   = allocF((size_t)BS_ * H_);
  float*    xsF  = allocF((size_t)BS_ * H_);
  uint16_t* hB   = allocH((size_t)BS_ * H_);
  uint16_t* qkvB = allocH((size_t)BS_ * 3 * H_);
  uint16_t* aoB  = allocH((size_t)BS_ * H_);
  uint16_t* cqB  = allocH((size_t)BS_ * H_);
  uint16_t* ckvB = allocH((size_t)BT_ * 2 * H_);
  uint16_t* ffB  = allocH((size_t)BS_ * FF_);
  uint16_t* xB   = allocH((size_t)BS_ * H_);

  auto cvt = [&](const float* s, uint16_t* d, long long n, float sc) {
    cvt_bf16_kernel<<<(unsigned)((n + 255) / 256), 256, 0, stream>>>(s, d, n, sc);
  };
  auto gemm = [&](const uint16_t* A, const uint16_t* W, const float* bias,
                  const float* resid, float* oF, uint16_t* oB,
                  int M, int N, int K, int act) {
    dim3 g((N + 127) / 128, (M + 127) / 128);
    gemm_bf16_wmma<<<g, 256, 0, stream>>>(A, W, bias, resid, oF, oB, M, N, K, act);
  };

  // One-time per launch: weights -> bf16, img*0.1 -> bf16, embedding -> fp32
  cvt(sqkv_w, wQKV, (long long)L_ * H_ * 3 * H_, 1.f);
  cvt(so_w,   wSO,  (long long)L_ * H_ * H_,     1.f);
  cvt(cq_w,   wCQ,  (long long)L_ * H_ * H_,     1.f);
  cvt(ckv_w,  wCKV, (long long)L_ * H_ * 2 * H_, 1.f);
  cvt(co_w,   wCO,  (long long)L_ * H_ * H_,     1.f);
  cvt(ff1_w,  wF1,  (long long)L_ * H_ * FF_,    1.f);
  cvt(ff2_w,  wF2,  (long long)L_ * FF_ * H_,    1.f);
  cvt(out_w,  wOUT, (long long)H_ * V_,          1.f);
  cvt(img,    imgB, (long long)BT_ * H_,         0.1f);
  embed_kernel<<<BS_ * H_ / 256, 256, 0, stream>>>(tok, tokEmb, posEmb, xF);

  const float iscale = 0.10206207261596577f;  // 1/sqrt(HD)
  const long long nAct = (long long)BS_ * H_;

  for (int l = 0; l < L_; ++l) {
    // xs = x * 0.1
    scale_f32_kernel<<<BS_ * H_ / 256, 256, 0, stream>>>(xF, xsF, nAct, 0.1f);
    // self-attention
    ln_kernel<<<BS_ / 4, 128, 0, stream>>>(xsF, ln1_g + (size_t)l * H_, ln1_b + (size_t)l * H_, hB, BS_);
    gemm(hB, wQKV + (size_t)l * H_ * 3 * H_, sqkv_b + (size_t)l * 3 * H_,
         nullptr, nullptr, qkvB, BS_, 3 * H_, H_, 0);
    attn_kernel<<<dim3(B_ * NH_, S_ / 8), 256, 0, stream>>>(
        qkvB, 3 * H_, qkvB, 3 * H_, H_, 2 * H_, aoB, S_, iscale);
    gemm(aoB, wSO + (size_t)l * H_ * H_, so_b + (size_t)l * H_,
         xsF, xsF, nullptr, BS_, H_, H_, 0);
    // cross-attention over image features
    ln_kernel<<<BS_ / 4, 128, 0, stream>>>(xsF, ln2_g + (size_t)l * H_, ln2_b + (size_t)l * H_, hB, BS_);
    gemm(hB, wCQ + (size_t)l * H_ * H_, cq_b + (size_t)l * H_,
         nullptr, nullptr, cqB, BS_, H_, H_, 0);
    gemm(imgB, wCKV + (size_t)l * H_ * 2 * H_, ckv_b + (size_t)l * 2 * H_,
         nullptr, nullptr, ckvB, BT_, 2 * H_, H_, 0);
    attn_kernel<<<dim3(B_ * NH_, S_ / 8), 256, 0, stream>>>(
        cqB, H_, ckvB, 2 * H_, 0, H_, aoB, T_, iscale);
    gemm(aoB, wCO + (size_t)l * H_ * H_, co_b + (size_t)l * H_,
         xsF, xsF, nullptr, BS_, H_, H_, 0);
    // FFN with exact GELU
    ln_kernel<<<BS_ / 4, 128, 0, stream>>>(xsF, ln3_g + (size_t)l * H_, ln3_b + (size_t)l * H_, hB, BS_);
    gemm(hB, wF1 + (size_t)l * H_ * FF_, ff1_b + (size_t)l * FF_,
         nullptr, nullptr, ffB, BS_, FF_, H_, 1);
    gemm(ffB, wF2 + (size_t)l * FF_ * H_, ff2_b + (size_t)l * H_,
         xsF, xsF, nullptr, BS_, H_, FF_, 0);
    // x = xs * 10
    scale_f32_kernel<<<BS_ * H_ / 256, 256, 0, stream>>>(xsF, xF, nAct, 10.f);
  }

  // Final vocab projection: [2048 x 768] @ [768 x 50257] + bias -> d_out (fp32)
  cvt(xF, xB, nAct, 1.f);
  gemm(xB, wOUT, out_b, nullptr, (float*)d_out, nullptr, BS_, V_, H_, 0);
}
